// PCgraph_67164698575226
// MI455X (gfx1250) — matrix operand order repaired and checked by reading, not compiled, not run on hardware
//
#include <hip/hip_runtime.h>
#include <hip/hip_bf16.h>
#include <math.h>

typedef __attribute__((ext_vector_type(16))) __bf16 v16bf;
typedef __attribute__((ext_vector_type(8)))  float  v8f;

#define N_NODES 4096
#define B_ROWS  32
#define NSENS   784
#define NLAB    10
#define LRX     0.1f

// float -> bf16 bits, round-to-nearest-even
__device__ __forceinline__ unsigned short f2bf(float f) {
    unsigned u = __float_as_uint(f);
    u += 0x7FFFu + ((u >> 16) & 1u);
    return (unsigned short)(u >> 16);
}

// ---------------- build dense A[s,t] = multiplicity * w[s,t] ----------------
__global__ void k_zero(float* a, int n) {
    int i = blockIdx.x * blockDim.x + threadIdx.x;
    if (i < n) a[i] = 0.f;
}

__global__ void k_scatter(const int* ei, const float* w, float* a, int E) {
    int e = blockIdx.x * blockDim.x + threadIdx.x;
    if (e >= E) return;
    int s = ei[e], t = ei[E + e];
    atomicAdd(&a[(size_t)s * N_NODES + t], w[(size_t)s * N_NODES + t]);
}

// Pack A (for err@A) and A^T (for f(x)@A^T) into WMMA-B lane-major tiles:
// tile (ntile,ktile) is 32 lanes x 8 dwords; lane's dword j holds bf16 pair
// (Bmat[k0+2j][n], Bmat[k0+2j+1][n]) with k0 = ktile*32 + (lane&16?16:0),
// n = ntile*16 + (lane&15).  One contiguous 32B read per lane per WMMA step.
__global__ void k_pack(const float* __restrict__ a,
                       unsigned* __restrict__ pA, unsigned* __restrict__ pAT) {
    unsigned tid = blockIdx.x * blockDim.x + threadIdx.x;
    bool isAT = tid >= (1u << 20);
    unsigned local = tid & ((1u << 20) - 1u);
    unsigned lane  = local & 31u;
    unsigned tpos  = local >> 5;
    unsigned ktile = tpos & 127u;
    unsigned ntile = tpos >> 7;
    unsigned n  = ntile * 16u + (lane & 15u);
    unsigned k0 = ktile * 32u + ((lane & 16u) ? 16u : 0u);
    unsigned* dst = (isAT ? pAT : pA) + (((ntile * 128u + ktile) * 32u + lane) * 8u);
#pragma unroll
    for (int j = 0; j < 8; ++j) {
        unsigned k = k0 + 2u * (unsigned)j;
        float lo, hi;
        if (isAT) { // Bmat[k=t][n=s] = A[s,t] = a[n*N + k]
            lo = a[(size_t)n * N_NODES + k];
            hi = a[(size_t)n * N_NODES + k + 1];
        } else {    // Bmat[k=s][n=t] = A[s,t] = a[k*N + n]
            lo = a[(size_t)k * N_NODES + n];
            hi = a[(size_t)(k + 1) * N_NODES + n];
        }
        dst[j] = ((unsigned)f2bf(hi) << 16) | (unsigned)f2bf(lo);
    }
}

// copy x -> working buffer (d_out) and seed Fh = bf16(tanh(x))
__global__ void k_init(const float* __restrict__ xin, float* __restrict__ xcur,
                       unsigned short* __restrict__ fh) {
    int i = blockIdx.x * blockDim.x + threadIdx.x;
    if (i < B_ROWS * N_NODES) {
        float v = xin[i];
        xcur[i] = v;
        fh[i] = f2bf(tanhf(v));
    }
}

// ---------------- fused GEMM + epilogue ----------------
// C[32 x 16] tile at column block ntile:  C = Asrc[32 x 4096] * Bpacked
// mode 0 (prediction): mu -> err = x - mu   (writes errf + err bf16)
// mode 1 (gradient):   g  -> aggr = tanh'(x)*g ; x -= lr*(err-aggr)*mask ;
//                      Fh = bf16(tanh(x_new))
__global__ void __launch_bounds__(256)
k_gemm(const unsigned short* __restrict__ asrc, const unsigned* __restrict__ bpk,
       float* __restrict__ xcur, float* __restrict__ errf,
       unsigned short* __restrict__ errh, unsigned short* __restrict__ fh,
       int mode) {
    __shared__ float red[8][32][8];
    const unsigned tid   = threadIdx.x;
    const unsigned lane  = tid & 31u;
    const unsigned wv    = tid >> 5;     // 8 waves
    const unsigned mt    = wv & 1u;      // M tile 0/1 (batch rows 0-15 / 16-31)
    const unsigned kch   = wv >> 1;      // K chunk 0..3 (1024 each)
    const unsigned ntile = blockIdx.x;   // 0..255

    // A-operand: 16-bit 16x32 layout -> two contiguous 16B chunks per lane
    const unsigned arow  = mt * 16u + (lane & 15u);
    const unsigned akoff = (lane & 16u) ? 16u : 0u;
    const unsigned short* aptr = asrc + (size_t)arow * N_NODES + kch * 1024u + akoff;
    // B-operand: pre-packed lane-major tiles, 2 x b128 per step
    const uint4* bptr = (const uint4*)(bpk +
        (((size_t)ntile * 128u + kch * 32u) * 32u + lane) * 8u);

    v8f acc = {};
    union { uint4 q[2]; v16bf v; } ua, ub;
#pragma unroll 4
    for (int kk = 0; kk < 32; ++kk) {              // 32 x K=32 -> K chunk of 1024
        ua.q[0] = *(const uint4*)(aptr);
        ua.q[1] = *(const uint4*)(aptr + 8);
        ub.q[0] = bptr[0];
        ub.q[1] = bptr[1];
        __builtin_prefetch(bptr + 64, 0, 1);       // next packed K tile (L2 hot)
        acc = __builtin_amdgcn_wmma_f32_16x16x32_bf16(
                  false, ua.v, false, ub.v, (short)0, acc, false, false);
        aptr += 32;   // K advances 32 bf16
        bptr += 64;   // 1 KB per packed tile
    }

#pragma unroll
    for (int r = 0; r < 8; ++r) red[wv][lane][r] = acc[r];
    __syncthreads();

    if (wv < 2) {   // one reducer wave per M tile
        float s[8];
#pragma unroll
        for (int r = 0; r < 8; ++r)
            s[r] = red[wv][lane][r] + red[wv + 2][lane][r] +
                   red[wv + 4][lane][r] + red[wv + 6][lane][r];
        const unsigned nloc  = lane & 15u;
        const unsigned mbase = wv * 16u + ((lane & 16u) ? 8u : 0u);
        const unsigned col   = ntile * 16u + nloc;
        const float msk = (col >= NSENS && col < (N_NODES - NLAB)) ? 1.f : 0.f;
#pragma unroll
        for (int r = 0; r < 8; ++r) {
            unsigned brow = mbase + (unsigned)r;
            size_t idx = (size_t)brow * N_NODES + col;
            if (mode == 0) {
                float xv = xcur[idx];
                float e  = xv - s[r];
                errf[idx] = e;
                errh[idx] = f2bf(e);
            } else {
                float xv   = xcur[idx];
                float th   = tanhf(xv);
                float aggr = (1.f - th * th) * s[r];
                float dEdx = errf[idx] - aggr;
                float xn   = xv - LRX * dEdx * msk;
                xcur[idx]  = xn;
                fh[idx]    = f2bf(tanhf(xn));
            }
        }
    }
}

extern "C" void kernel_launch(void* const* d_in, const int* in_sizes, int n_in,
                              void* d_out, int out_size, void* d_ws, size_t ws_size,
                              hipStream_t stream) {
    const float* x  = (const float*)d_in[0];
    const float* w  = (const float*)d_in[1];
    const int*   ei = (const int*)d_in[2];
    const int    E  = in_sizes[2] / 2;
    float* xcur = (float*)d_out;                 // working x, becomes x_final

    char* ws = (char*)d_ws;
    float*          Af  = (float*)ws;          ws += (size_t)N_NODES * N_NODES * 4; // 64 MB
    unsigned*       pA  = (unsigned*)ws;       ws += (size_t)N_NODES * N_NODES * 2; // 32 MB
    unsigned*       pAT = (unsigned*)ws;       ws += (size_t)N_NODES * N_NODES * 2; // 32 MB
    unsigned short* fh  = (unsigned short*)ws; ws += (size_t)B_ROWS * N_NODES * 2;
    unsigned short* eh  = (unsigned short*)ws; ws += (size_t)B_ROWS * N_NODES * 2;
    float*          ef  = (float*)ws;

    const int NN = N_NODES * N_NODES;
    k_zero<<<(NN + 255) / 256, 256, 0, stream>>>(Af, NN);
    k_scatter<<<(E + 255) / 256, 256, 0, stream>>>(ei, w, Af, E);
    k_pack<<<(2u << 20) / 256, 256, 0, stream>>>(Af, pA, pAT);
    k_init<<<(B_ROWS * N_NODES + 255) / 256, 256, 0, stream>>>(x, xcur, fh);

    for (int t = 0; t < 6; ++t) {
        // mu = tanh(x) @ A^T ; err = x - mu
        k_gemm<<<256, 256, 0, stream>>>(fh, pAT, xcur, ef, eh, fh, 0);
        // g = err @ A ; x -= lr*(err - tanh'(x)*g)*mask ; refresh tanh(x)
        k_gemm<<<256, 256, 0, stream>>>(eh, pA, xcur, ef, eh, fh, 1);
    }
    (void)n_in; (void)out_size; (void)ws_size;
}
